// GCN_net_53970559042213
// MI455X (gfx1250) — matrix operand order repaired and checked by reading, not compiled, not run on hardware
//
#include <hip/hip_runtime.h>

typedef __bf16 bf16;
typedef __attribute__((ext_vector_type(16))) __bf16 v16bf;
typedef __attribute__((ext_vector_type(8)))  float  v8f;

static constexpr int NN  = 100000;   // nodes
static constexpr int EE  = 1600000;  // edges
static constexpr int FIN = 100;      // input features
static constexpr int D   = 64;       // hidden dim (D1 == D2)
static constexpr int KP  = 128;      // FIN padded to multiple of 32
static constexpr int GP  = 512;      // graphs padded to multiple of 16 (G = 500)

// ---------------------------------------------------------------- edge prep
__global__ void k_deg(const int* __restrict__ ei, const float* __restrict__ w,
                      float* __restrict__ deg, int E) {
  int e = blockIdx.x * 256 + threadIdx.x;
  if (e < E) atomicAdd(&deg[ei[E + e]], w[e]);   // segment over col
}

__global__ void k_dis(float* __restrict__ deg, int n) {
  int i = blockIdx.x * 256 + threadIdx.x;
  if (i < n) { float d = deg[i]; deg[i] = (d > 0.f) ? rsqrtf(d) : 0.f; }
}

__global__ void k_norm(const int* __restrict__ ei, const float* __restrict__ w,
                       const float* __restrict__ dis, float* __restrict__ norm, int E) {
  int e = blockIdx.x * 256 + threadIdx.x;
  if (e < E) norm[e] = dis[ei[e]] * w[e] * dis[ei[E + e]];
}

// ---------------------------------------------------------------- conversions
__global__ void k_cvt_x(const float* __restrict__ x, bf16* __restrict__ xb, int n) {
  int i = blockIdx.x * 256 + threadIdx.x;              // n = NN*KP
  if (i < n) {
    int r = i >> 7, k = i & (KP - 1);
    xb[i] = (k < FIN) ? (bf16)x[r * FIN + k] : (bf16)0.f;
  }
}

__global__ void k_cvt_w0(const float* __restrict__ w, bf16* __restrict__ wb) {
  int i = blockIdx.x * 256 + threadIdx.x;              // KP*D
  if (i < KP * D) {
    int k = i >> 6, j = i & 63;
    wb[i] = (k < FIN) ? (bf16)w[k * D + j] : (bf16)0.f;
  }
}

__global__ void k_cvt(const float* __restrict__ s, bf16* __restrict__ d, int n) {
  int i = blockIdx.x * 256 + threadIdx.x;
  if (i < n) d[i] = (bf16)s[i];
}

// ---------------------------------------------------------------- WMMA GEMM
// C[M x 64] = A[M x K](bf16) @ B[K x 64](bf16); K = S*32, M % 16 == 0.
// mode 1: Cb = bf16(relu(acc + bias));  mode 2: Cf = acc (raw fp32, no bias).
// B is staged once into LDS as WMMA fragments, then hoisted into registers
// (loop-invariant, S*4 v16bf = up to 128 VGPRs). Each wave then sweeps TPW
// 16-row tiles: hot loop = 2x global_load_b128 (A) + 4x v_wmma, no LDS.
template<int S, int TPW>
__global__ void __launch_bounds__(256)
k_gemm(const bf16* __restrict__ A, const bf16* __restrict__ B,
       const float* __restrict__ bias, bf16* __restrict__ Cb,
       float* __restrict__ Cf, int M, int mode) {
  constexpr int K = S * 32;
  __shared__ unsigned ldsB[S * 1024];
  const int tid  = threadIdx.x;
  const int lane = tid & 31;
  const int wave = tid >> 5;

  // Stage B as pre-swizzled WMMA fragments: frag(s,t) lane L dword v holds
  // packed bf16 pair {K = s*32 + (L/16)*8 + 2v + (v/4)*8, +1} at column N = t*16 + L%16.
  for (int i = tid; i < S * 1024; i += 256) {
    int v = i & 7, L = (i >> 3) & 31, t = (i >> 8) & 3, s = i >> 10;
    int n  = (L & 15) + t * 16;
    int k0 = (s << 5) + ((L >> 4) << 3) + (v << 1) + ((v >> 2) << 3);
    union { bf16 h[2]; unsigned u; } p;
    p.h[0] = B[k0 * D + n];
    p.h[1] = B[(k0 + 1) * D + n];
    ldsB[i] = p.u;
  }
  __syncthreads();

  // Hoist all B fragments to registers (invariant across tiles).
  v16bf bfrag[S * 4];
  const v16bf* bbase = ((const v16bf*)ldsB) + lane;
#pragma unroll
  for (int s = 0; s < S; ++s)
#pragma unroll
    for (int t = 0; t < 4; ++t)
      bfrag[s * 4 + t] = bbase[s * 128 + t * 32];

  const int m  = lane & 15;                            // row within tile
  const int kh = lane >> 4;                            // K-half select
  const int ntiles = M >> 4;

#pragma unroll
  for (int j = 0; j < TPW; ++j) {
    const int tile = (blockIdx.x * 8 + wave) * TPW + j;    // wave-uniform
    if (tile >= ntiles) break;                             // whole wave: EXEC stays full

    const bf16* arow = A + ((size_t)tile * 16 + m) * K + kh * 8;
    if (j + 1 < TPW && tile + 1 < ntiles)
      __builtin_prefetch(arow + (size_t)16 * K, 0, 3);

    v8f z = {0.f, 0.f, 0.f, 0.f, 0.f, 0.f, 0.f, 0.f};
    v8f acc[4] = {z, z, z, z};
#pragma unroll
    for (int s = 0; s < S; ++s) {
      union { uint4 q[2]; v16bf v; } a;
      a.q[0] = *(const uint4*)(arow + s * 32);           // K = s*32 + kh*8 .. +7
      a.q[1] = *(const uint4*)(arow + s * 32 + 16);      // K = s*32 + 16 + kh*8 .. +7
#pragma unroll
      for (int t = 0; t < 4; ++t)
        acc[t] = __builtin_amdgcn_wmma_f32_16x16x32_bf16(
            false, a.v, false, bfrag[s * 4 + t], (short)0, acc[t], false, false);
    }

    // C layout: dword r -> row = tile*16 + (lane/16)*8 + r, col = t*16 + lane%16
    const int rbase = tile * 16 + kh * 8;
    if (mode == 1) {
#pragma unroll
      for (int t = 0; t < 4; ++t) {
        int c = t * 16 + m;
        float bv = bias[c];
#pragma unroll
        for (int r = 0; r < 8; ++r) {
          float v = acc[t][r] + bv;
          Cb[(size_t)(rbase + r) * D + c] = (bf16)(v > 0.f ? v : 0.f);
        }
      }
    } else {
#pragma unroll
      for (int t = 0; t < 4; ++t) {
        int c = t * 16 + m;
#pragma unroll
        for (int r = 0; r < 8; ++r)
          Cf[(size_t)(rbase + r) * D + c] = acc[t][r];
      }
    }
  }
}

// ---------------------------------------------------------------- edge message pass
__global__ void k_scatter(const float* __restrict__ h, const float* __restrict__ norm,
                          const int* __restrict__ ei, float* __restrict__ agg, int E) {
  int i = blockIdx.x * 256 + threadIdx.x;              // E * 16 threads
  int e = i >> 4, c = i & 15;
  if (e < E) {
    float nw = norm[e];
    int r = ei[e], cl = ei[E + e];
    float4 hv = *(const float4*)(h + (size_t)r * D + c * 4);
    float* dst = agg + (size_t)cl * D + c * 4;
    atomicAdd(dst + 0, hv.x * nw);
    atomicAdd(dst + 1, hv.y * nw);
    atomicAdd(dst + 2, hv.z * nw);
    atomicAdd(dst + 3, hv.w * nw);
  }
}

__global__ void k_bias_relu(const float* __restrict__ agg, const float* __restrict__ b,
                            bf16* __restrict__ outb, int n) {
  int i = blockIdx.x * 256 + threadIdx.x;              // n = NN*D
  if (i < n) {
    float v = agg[i] + b[i & 63];
    outb[i] = (bf16)(v > 0.f ? v : 0.f);
  }
}

// ---------------------------------------------------------------- pooling + head
__global__ void k_pool(const bf16* __restrict__ outb, const int* __restrict__ batch,
                       float* __restrict__ sums, float* __restrict__ cnt, int n) {
  int i = blockIdx.x * 256 + threadIdx.x;              // n = NN*D
  if (i < n) {
    int nd = i >> 6, d = i & 63;
    int g = batch[nd];
    atomicAdd(&sums[g * D + d], (float)outb[i]);
    if (d == 0) atomicAdd(&cnt[g], 1.f);
  }
}

__global__ void k_poolfin(const float* __restrict__ sums, const float* __restrict__ cnt,
                          bf16* __restrict__ pb, int G) {
  int i = blockIdx.x * 256 + threadIdx.x;              // GP*D
  if (i < GP * D) {
    int g = i >> 6;
    float v = 0.f;
    if (g < G) v = sums[i] / fmaxf(cnt[g], 1.f);
    pb[i] = (bf16)v;
  }
}

__global__ void k_lin2(const bf16* __restrict__ a, const float* __restrict__ w,
                       const float* __restrict__ b, float* __restrict__ out, int G) {
  int g = blockIdx.x * 256 + threadIdx.x;
  if (g < G) {
    float s = 0.f;
    for (int d = 0; d < D; ++d) s += (float)a[g * D + d] * w[d];
    out[g] = s + b[0];
  }
}

// ---------------------------------------------------------------- launcher
extern "C" void kernel_launch(void* const* d_in, const int* in_sizes, int n_in,
                              void* d_out, int out_size, void* d_ws, size_t ws_size,
                              hipStream_t stream) {
  const float* x       = (const float*)d_in[0];
  const int*   ei      = (const int*)  d_in[1];
  const float* ew      = (const float*)d_in[2];
  const int*   batch   = (const int*)  d_in[3];
  const float* lin0_w  = (const float*)d_in[4];
  const float* lin0_b  = (const float*)d_in[5];
  const float* conv_w  = (const float*)d_in[6];
  const float* conv_b  = (const float*)d_in[7];
  const float* lin1_w  = (const float*)d_in[8];
  const float* lin1_b  = (const float*)d_in[9];
  const float* fc_w    = (const float*)d_in[10];
  const float* fc_b    = (const float*)d_in[11];
  const float* lin2_w  = (const float*)d_in[12];
  const float* lin2_b  = (const float*)d_in[13];
  float* out = (float*)d_out;
  const int G = out_size;                              // 500

  // workspace carve-out (256B aligned)
  char* p = (char*)d_ws;
  auto carve = [&](size_t bytes) { void* r = p; p += (bytes + 255) & ~(size_t)255; return r; };
  float* dis   = (float*)carve((size_t)NN * 4);        // deg -> deg^{-1/2} in place
  float* norm  = (float*)carve((size_t)EE * 4);
  bf16*  xb    = (bf16*) carve((size_t)NN * KP * 2);
  bf16*  w0b   = (bf16*) carve((size_t)KP * D * 2);
  bf16*  cwb   = (bf16*) carve((size_t)4 * D * D * 2);
  bf16*  l1wb  = (bf16*) carve((size_t)D * D * 2);
  bf16*  fcwb  = (bf16*) carve((size_t)2 * D * D * 2);
  bf16*  outb  = (bf16*) carve((size_t)NN * D * 2);
  float* h     = (float*)carve((size_t)NN * D * 4);
  float* agg   = (float*)carve((size_t)NN * D * 4);
  float* sums  = (float*)carve((size_t)GP * D * 4);
  float* cnt   = (float*)carve((size_t)GP * 4);
  bf16*  pb0   = (bf16*) carve((size_t)GP * D * 2);
  bf16*  pb1   = (bf16*) carve((size_t)GP * D * 2);

  auto blks = [](long long n) { return (unsigned)((n + 255) / 256); };

  // 1) GCN normalization
  hipMemsetAsync(dis, 0, (size_t)NN * 4, stream);
  k_deg <<<blks(EE), 256, 0, stream>>>(ei, ew, dis, EE);
  k_dis <<<blks(NN), 256, 0, stream>>>(dis, NN);
  k_norm<<<blks(EE), 256, 0, stream>>>(ei, ew, dis, norm, EE);

  // 2) bf16 conversions
  k_cvt_x <<<blks((long long)NN * KP), 256, 0, stream>>>(x, xb, NN * KP);
  k_cvt_w0<<<blks(KP * D), 256, 0, stream>>>(lin0_w, w0b);
  k_cvt   <<<blks(4 * D * D), 256, 0, stream>>>(conv_w, cwb, 4 * D * D);
  k_cvt   <<<blks(D * D), 256, 0, stream>>>(lin1_w, l1wb, D * D);
  k_cvt   <<<blks(2 * D * D), 256, 0, stream>>>(fc_w, fcwb, 2 * D * D);

  // GEMM grids: block = 8 waves x 4 tiles = 512 rows
  const unsigned gemmN = (unsigned)((NN / 16 + 31) / 32);  // 196 blocks
  const unsigned gemmG = (GP / 16 + 31) / 32;              // 1 block

  // 3) lin0: out = relu(x @ W0 + b0)   [WMMA bf16, K=128]
  k_gemm<4, 4><<<gemmN, 256, 0, stream>>>(xb, w0b, lin0_b, outb, nullptr, NN, 1);

  // 4) conv layers
  for (int i = 0; i < 4; ++i) {
    k_gemm<2, 4><<<gemmN, 256, 0, stream>>>(outb, cwb + i * D * D, nullptr, nullptr, h, NN, 2);
    hipMemsetAsync(agg, 0, (size_t)NN * D * 4, stream);
    k_scatter<<<blks((long long)EE * 16), 256, 0, stream>>>(h, norm, ei, agg, EE);
    k_bias_relu<<<blks((long long)NN * D), 256, 0, stream>>>(agg, conv_b + i * D, outb, NN * D);
  }

  // 5) global mean pool
  hipMemsetAsync(sums, 0, (size_t)GP * D * 4, stream);
  hipMemsetAsync(cnt,  0, (size_t)GP * 4, stream);
  k_pool   <<<blks((long long)NN * D), 256, 0, stream>>>(outb, batch, sums, cnt, NN * D);
  k_poolfin<<<blks(GP * D), 256, 0, stream>>>(sums, cnt, pb0, G);

  // 6) head: lin1 + 2 fc + lin2   [WMMA bf16, M=512]
  k_gemm<2, 4><<<gemmG, 256, 0, stream>>>(pb0, l1wb, lin1_b, pb1, nullptr, GP, 1);
  k_gemm<2, 4><<<gemmG, 256, 0, stream>>>(pb1, fcwb,         fc_b,     pb0, nullptr, GP, 1);
  k_gemm<2, 4><<<gemmG, 256, 0, stream>>>(pb0, fcwb + D * D, fc_b + D, pb1, nullptr, GP, 1);
  k_lin2<<<blks(G), 256, 0, stream>>>(pb1, lin2_w, lin2_b, out, G);
}
